// TrainableMMSEModule_60430189854928
// MI455X (gfx1250) — compile-verified, matrix operand order checked by reading, not run on hardware
//
#include <hip/hip_runtime.h>
#include <hip/hip_bf16.h>

typedef __attribute__((ext_vector_type(16))) _Float16 v16h;
typedef __attribute__((ext_vector_type(8)))  float    v8f;

#define WMMA16(a, b, c) __builtin_amdgcn_wmma_f32_16x16x32_f16( \
    false, (a), false, (b), (short)0, (c), false, false)

// ---- packed weight layout in d_ws (f16 elements) ----
// Each tile = 32 lanes * 16 halves = 512 f16, element (lane,j) holds
// B[K = ks*32 + j + (j&8) + 8*(lane>=16)][N = nt*16 + (lane&15)]
static constexpr int TILE_E   = 512;
static constexpr int OFF_B1   = 0;                      // 1 ks * 8 nt  = 8 tiles
static constexpr int OFF_B2   = OFF_B1 + 8  * TILE_E;   // 4*8 = 32 tiles
static constexpr int OFF_B3   = OFF_B2 + 32 * TILE_E;   // 4*4 = 16 tiles
static constexpr int OFF_B4   = OFF_B3 + 16 * TILE_E;   // 2*9 = 18 tiles
static constexpr int MLP_STRIDE = OFF_B4 + 18 * TILE_E; // 37888 f16 per MLP

struct MlpParams {
    const _Float16 *B1, *B2, *B3, *B4;
    const float *b1, *g1, *be1, *b2, *g2, *be2, *b3, *b4;
};

// ---------------- weight packing kernel ----------------
__global__ void pack_w_kernel(const float* __restrict__ W, _Float16* __restrict__ dst,
                              int din, int dout, int NT) {
    int tile = blockIdx.x;           // ks*NT + nt
    int ks = tile / NT, nt = tile % NT;
    int t = threadIdx.x;             // 0..511
    int lane = t >> 4, j = t & 15;
    int K = ks * 32 + j + (j & 8) + ((lane >> 4) << 3);
    int N = nt * 16 + (lane & 15);
    float v = (K < din && N < dout) ? W[(size_t)K * dout + N] : 0.0f;
    dst[(size_t)tile * TILE_E + lane * 16 + j] = (_Float16)v;
}

// ---------------- device helpers ----------------
__device__ __forceinline__ float lrelu(float z) { return z >= 0.0f ? z : 0.1f * z; }

__device__ __forceinline__ void ln_stats(int lane, const float* __restrict__ hbuf,
                                         float* rowm, float* rowr) {
    if (lane < 16) {
        const float* p = hbuf + lane * 128;
        float s = 0.0f, ss = 0.0f;
        #pragma unroll
        for (int c = 0; c < 128; c += 4) {
            float4 q = *(const float4*)(p + c);
            s  += q.x + q.y + q.z + q.w;
            ss += q.x * q.x + q.y * q.y + q.z * q.z + q.w * q.w;
        }
        float mn  = s * (1.0f / 128.0f);
        float var = ss * (1.0f / 128.0f) - mn * mn;
        rowm[lane] = mn;
        rowr[lane] = rsqrtf(var + 1e-5f);
    }
}

__device__ __forceinline__ void run_mlp(int lane, const v16h a1, const MlpParams P,
                                        float* hbuf, float* fbuf,
                                        float* rowm, float* rowr) {
    const int ncolidx = lane & 15;
    const int mhi     = (lane >> 4) * 8;
    const int khi     = (lane >> 4) << 3;
    const int row     = lane & 15;

    // ---- Layer 1: [16x32(f16)] @ [32x128]  (K padded 24->32) ----
    #pragma unroll
    for (int nt = 0; nt < 8; ++nt) {
        v16h b = *(const v16h*)(P.B1 + (size_t)nt * TILE_E + lane * 16);
        v8f c = {};
        c = WMMA16(a1, b, c);
        float bias = P.b1[nt * 16 + ncolidx];
        #pragma unroll
        for (int v = 0; v < 8; ++v)
            hbuf[(mhi + v) * 128 + nt * 16 + ncolidx] = c[v] + bias;
    }
    __syncthreads();
    ln_stats(lane, hbuf, rowm, rowr);
    __syncthreads();

    v16h a2[4];
    {
        float mn = rowm[row], rs = rowr[row];
        #pragma unroll
        for (int ks = 0; ks < 4; ++ks) {
            v16h a;
            #pragma unroll
            for (int j = 0; j < 16; ++j) {
                int K = ks * 32 + j + (j & 8) + khi;
                float y = (hbuf[row * 128 + K] - mn) * rs * P.g1[K] + P.be1[K];
                a[j] = (_Float16)lrelu(y);
            }
            a2[ks] = a;
        }
    }
    __syncthreads();

    // ---- Layer 2: [16x128] @ [128x128] ----
    #pragma unroll
    for (int nt = 0; nt < 8; ++nt) {
        v8f c = {};
        #pragma unroll
        for (int ks = 0; ks < 4; ++ks) {
            v16h b = *(const v16h*)(P.B2 + (size_t)(ks * 8 + nt) * TILE_E + lane * 16);
            c = WMMA16(a2[ks], b, c);
        }
        float bias = P.b2[nt * 16 + ncolidx];
        #pragma unroll
        for (int v = 0; v < 8; ++v)
            hbuf[(mhi + v) * 128 + nt * 16 + ncolidx] = c[v] + bias;
    }
    __syncthreads();
    ln_stats(lane, hbuf, rowm, rowr);
    __syncthreads();

    v16h a3[4];
    {
        float mn = rowm[row], rs = rowr[row];
        #pragma unroll
        for (int ks = 0; ks < 4; ++ks) {
            v16h a;
            #pragma unroll
            for (int j = 0; j < 16; ++j) {
                int K = ks * 32 + j + (j & 8) + khi;
                float y = (hbuf[row * 128 + K] - mn) * rs * P.g2[K] + P.be2[K];
                a[j] = (_Float16)lrelu(y);
            }
            a3[ks] = a;
        }
    }
    __syncthreads();

    // ---- Layer 3: [16x128] @ [128x64], bias + lrelu ----
    #pragma unroll
    for (int nt = 0; nt < 4; ++nt) {
        v8f c = {};
        #pragma unroll
        for (int ks = 0; ks < 4; ++ks) {
            v16h b = *(const v16h*)(P.B3 + (size_t)(ks * 4 + nt) * TILE_E + lane * 16);
            c = WMMA16(a3[ks], b, c);
        }
        float bias = P.b3[nt * 16 + ncolidx];
        #pragma unroll
        for (int v = 0; v < 8; ++v)
            hbuf[(mhi + v) * 64 + nt * 16 + ncolidx] = lrelu(c[v] + bias);
    }
    __syncthreads();

    v16h a4[2];
    #pragma unroll
    for (int ks = 0; ks < 2; ++ks) {
        v16h a;
        #pragma unroll
        for (int j = 0; j < 16; ++j) {
            int K = ks * 32 + j + (j & 8) + khi;
            a[j] = (_Float16)hbuf[row * 64 + K];
        }
        a4[ks] = a;
    }
    __syncthreads();

    // ---- Layer 4: [16x64] @ [64x144] -> fbuf ----
    #pragma unroll
    for (int nt = 0; nt < 9; ++nt) {
        v8f c = {};
        #pragma unroll
        for (int ks = 0; ks < 2; ++ks) {
            v16h b = *(const v16h*)(P.B4 + (size_t)(ks * 9 + nt) * TILE_E + lane * 16);
            c = WMMA16(a4[ks], b, c);
        }
        float bias = P.b4[nt * 16 + ncolidx];
        #pragma unroll
        for (int v = 0; v < 8; ++v)
            fbuf[(mhi + v) * 144 + nt * 16 + ncolidx] = c[v] + bias;
    }
}

// ---------------- main kernel: 1 wave / block, 16 rows / wave ----------------
__global__ __launch_bounds__(32)
void mmse_mlp_kernel(const float* __restrict__ xr, const float* __restrict__ xi,
                     MlpParams pc, MlpParams pr, float* __restrict__ out, int Mrows) {
    __shared__ float hbuf[16 * 128];
    __shared__ float fbuf[16 * 144];
    __shared__ float rowm[16];
    __shared__ float rowr[16];

    const int lane = threadIdx.x;
    const int r0   = blockIdx.x * 16;

    // Build A1 fragment: 16 rows x 32 K (K 0..11 real, 12..23 imag, 24..31 zero)
    v16h a1;
    {
        const int row = r0 + (lane & 15);
        const int khi = (lane >> 4) << 3;
        #pragma unroll
        for (int j = 0; j < 16; ++j) {
            int K = j + (j & 8) + khi;
            float v = 0.0f;
            if (K < 12)      v = xr[(size_t)row * 12 + K];
            else if (K < 24) v = xi[(size_t)row * 12 + (K - 12)];
            a1[j] = (_Float16)v;
        }
    }

    #pragma unroll
    for (int which = 0; which < 2; ++which) {
        run_mlp(lane, a1, which ? pr : pc, hbuf, fbuf, rowm, rowr);
        __syncthreads();

        // Scatter factors -> (Lr, Li); each lane builds 4 consecutive elements
        // so the stores are aligned global_store_b128 (row = 144 f32 = 36 float4).
        const size_t base_r = (size_t)(2 * which) * Mrows * 144;
        const size_t base_i = base_r + (size_t)Mrows * 144;
        for (int r = 0; r < 16; ++r) {
            const float* f = &fbuf[r * 144];
            const size_t rb = (size_t)(r0 + r) * 144;
            for (int g = lane; g < 36; g += 32) {
                const int e0 = g * 4;
                float4 vr, vi;
                float* vrp = &vr.x;
                float* vip = &vi.x;
                #pragma unroll
                for (int q = 0; q < 4; ++q) {
                    int e = e0 + q;
                    int i = e / 12, j = e % 12;
                    float a, b;
                    if (i == j)     { a = f[i];             b = 0.0f; }
                    else if (i > j) { int t = (i * (i - 1)) / 2 + j;
                                      a = f[12 + t];        b = f[78 + t]; }
                    else            { a = 0.0f;             b = 0.0f; }
                    vrp[q] = a;
                    vip[q] = b;
                }
                *(float4*)(out + base_r + rb + e0) = vr;
                *(float4*)(out + base_i + rb + e0) = vi;
            }
        }
        __syncthreads();   // fbuf reused by the next MLP
    }
}

extern "C" void kernel_launch(void* const* d_in, const int* in_sizes, int n_in,
                              void* d_out, int out_size, void* d_ws, size_t ws_size,
                              hipStream_t stream) {
    const float* xr = (const float*)d_in[0];
    const float* xi = (const float*)d_in[1];
    const int Mrows = in_sizes[0] / 12;

    _Float16* ws = (_Float16*)d_ws;
    _Float16* wsC = ws;
    _Float16* wsR = ws + MLP_STRIDE;

    // Pack weights (C at d_in[2..13], R at d_in[14..25]).
    for (int m = 0; m < 2; ++m) {
        const int base = 2 + 12 * m;
        _Float16* dst = m ? wsR : wsC;
        pack_w_kernel<<< 8, 512, 0, stream>>>((const float*)d_in[base + 0], dst + OFF_B1,  24, 128, 8);
        pack_w_kernel<<<32, 512, 0, stream>>>((const float*)d_in[base + 4], dst + OFF_B2, 128, 128, 8);
        pack_w_kernel<<<16, 512, 0, stream>>>((const float*)d_in[base + 8], dst + OFF_B3, 128,  64, 4);
        pack_w_kernel<<<18, 512, 0, stream>>>((const float*)d_in[base +10], dst + OFF_B4,  64, 144, 9);
    }

    MlpParams pc {
        wsC + OFF_B1, wsC + OFF_B2, wsC + OFF_B3, wsC + OFF_B4,
        (const float*)d_in[3],  (const float*)d_in[4],  (const float*)d_in[5],
        (const float*)d_in[7],  (const float*)d_in[8],  (const float*)d_in[9],
        (const float*)d_in[11], (const float*)d_in[13]
    };
    MlpParams pr {
        wsR + OFF_B1, wsR + OFF_B2, wsR + OFF_B3, wsR + OFF_B4,
        (const float*)d_in[15], (const float*)d_in[16], (const float*)d_in[17],
        (const float*)d_in[19], (const float*)d_in[20], (const float*)d_in[21],
        (const float*)d_in[23], (const float*)d_in[25]
    };

    const int nblocks = Mrows / 16;   // 16384 row-tiles
    mmse_mlp_kernel<<<nblocks, 32, 0, stream>>>(xr, xi, pc, pr, (float*)d_out, Mrows);
}
